// BAKT_QIKT_1365799600740
// MI455X (gfx1250) — matrix operand order repaired and checked by reading, not compiled, not run on hardware
//
#include <hip/hip_runtime.h>
#include <hip/hip_bf16.h>
#include <math.h>

#define BSZ 16
#define NH  8
#define SEQ 1024
#define DKD 128
#define NEGV -1e32f
#define KMAX 8
#define PSTR (SEQ + 8)   // padded bf16 row stride (break LDS bank conflicts)

typedef __attribute__((ext_vector_type(16))) __bf16 v16bf;
typedef __attribute__((ext_vector_type(8)))  float  v8f;

__device__ __forceinline__ float wave_max(float v) {
#pragma unroll
  for (int o = 16; o; o >>= 1) v = fmaxf(v, __shfl_xor(v, o, 32));
  return v;
}
__device__ __forceinline__ float wave_sum(float v) {
#pragma unroll
  for (int o = 16; o; o >>= 1) v += __shfl_xor(v, o, 32);
  return v;
}

__global__ __launch_bounds__(256)
void topk_causal_attn(const float* __restrict__ Q,
                      const float* __restrict__ K,
                      const float* __restrict__ V,
                      const int*   __restrict__ kidx_p,
                      float*       __restrict__ O)
{
  extern __shared__ char smem[];
  float*  sS = (float*)smem;                                  // [16][SEQ] f32 scores
  __bf16* sP = (__bf16*)(smem + 16 * SEQ * sizeof(float));    // [16][PSTR] bf16 probs

  const int qt   = blockIdx.x;          // 16-row q tile index
  const int bh   = blockIdx.y;          // flattened batch*head
  const int q0   = qt * 16;
  const int tid  = threadIdx.x;
  const int w    = tid >> 5;            // wave id (0..7)
  const int lane = tid & 31;
  const int half = lane >> 4;           // lane half for WMMA K-striping
  const int r    = lane & 15;           // row/col within 16

  const size_t base = (size_t)bh * SEQ * DKD;
  const float* Qp = Q + base;
  const float* Kp = K + base;
  const float* Vp = V + base;
  float*       Op = O + base;

  int kidx = *kidx_p;
  if (kidx < 1) kidx = 1;
  if (kidx > KMAX) kidx = KMAX;

  const float scale = 1.0f / sqrtf((float)DKD);

  // ---------------- Phase 1: S = Q K^T (bf16 WMMA, causal-skipped) -----------
  // Preload Q-tile A operand: 4 chunks of 16x32 bf16.
  // A layout (16-bit 16x32): lanes 0-15 -> M=lane, K={0..7}u{16..23};
  //                          lanes 16-31 -> M=lane-16, K={8..15}u{24..31}.
  v16bf aq[4];
#pragma unroll
  for (int kk = 0; kk < 4; ++kk) {
    const float* qrow = Qp + (size_t)(q0 + r) * DKD + kk * 32 + half * 8;
    float4 f0 = *(const float4*)(qrow + 0);
    float4 f1 = *(const float4*)(qrow + 4);
    float4 f2 = *(const float4*)(qrow + 16);
    float4 f3 = *(const float4*)(qrow + 20);
    float t[16] = { f0.x, f0.y, f0.z, f0.w, f1.x, f1.y, f1.z, f1.w,
                    f2.x, f2.y, f2.z, f2.w, f3.x, f3.y, f3.z, f3.w };
#pragma unroll
    for (int j = 0; j < 16; ++j) aq[kk][j] = (__bf16)t[j];
  }

  for (int nb = w; nb <= qt; nb += 8) {     // only blocks touching the causal region
    const int n0 = nb * 16;
    v8f acc = {};
#pragma unroll
    for (int kk = 0; kk < 4; ++kk) {
      // B layout (16-bit 32x16): lane -> N = lane&15; lanes 0-15 hold K={0..7}u{16..23},
      // lanes 16-31 hold K={8..15}u{24..31} (same K order as A).
      const float* krow = Kp + (size_t)(n0 + r) * DKD + kk * 32 + half * 8;
      float4 g0 = *(const float4*)(krow + 0);
      float4 g1 = *(const float4*)(krow + 4);
      float4 g2 = *(const float4*)(krow + 16);
      float4 g3 = *(const float4*)(krow + 20);
      float t[16] = { g0.x, g0.y, g0.z, g0.w, g1.x, g1.y, g1.z, g1.w,
                      g2.x, g2.y, g2.z, g2.w, g3.x, g3.y, g3.z, g3.w };
      v16bf b;
#pragma unroll
      for (int j = 0; j < 16; ++j) b[j] = (__bf16)t[j];
      acc = __builtin_amdgcn_wmma_f32_16x16x32_bf16(
          false, aq[kk], false, b, (short)0, acc, false, false);
    }
    // C/D layout: lane -> N = lane&15; VGPR i -> M = i + 8*(lane>>4)
#pragma unroll
    for (int i = 0; i < 8; ++i) {
      const int M  = i + 8 * half;
      const int n  = n0 + r;
      const int qg = q0 + M;
      sS[M * SEQ + n] = (n <= qg) ? acc[i] * scale : NEGV;
    }
  }
  __syncthreads();

  // ------------- Phase 2: softmax -> top-k threshold -> re-softmax -----------
  const int Lv   = (qt + 1) * 16;          // valid (causal) column extent
  const int Lv32 = (Lv + 31) & ~31;        // rounded to WMMA k-chunk

  for (int rr = 0; rr < 2; ++rr) {
    const int M   = w + rr * 8;            // one wave per row, 2 rows/wave
    const int qg  = q0 + M;
    const int len = qg + 1;
    const float* srow = sS + M * SEQ;

    float m = -3.0e38f;
    for (int n = lane; n < len; n += 32) m = fmaxf(m, srow[n]);
    const float m1 = wave_max(m);

    float s1 = 0.f;
    for (int n = lane; n < len; n += 32) s1 += __expf(srow[n] - m1);
    const float sum1 = wave_sum(s1);
    const float inv1 = 1.0f / sum1;

    // k-th largest score (monotone in prob) via per-lane sorted top-8 + shuffle merge
    float kth = -3.0e38f;
    if (qg >= kidx) {
      float t[KMAX];
#pragma unroll
      for (int j = 0; j < KMAX; ++j) t[j] = -3.0e38f;
      for (int n = lane; n < len; n += 32) {
        float v = srow[n];
#pragma unroll
        for (int j = 0; j < KMAX; ++j) {   // bubble-insert, keeps t sorted desc
          float hi = fmaxf(t[j], v);
          float lo = fminf(t[j], v);
          t[j] = hi; v = lo;
        }
      }
#pragma unroll
      for (int off = 16; off; off >>= 1) {
        float o[KMAX];
#pragma unroll
        for (int j = 0; j < KMAX; ++j) o[j] = __shfl_xor(t[j], off, 32);
        float mg[KMAX];
        int ia = 0, ib = 0;
#pragma unroll
        for (int j = 0; j < KMAX; ++j) {   // merge two sorted-desc lists, keep top KMAX
          float va = (ia < KMAX) ? t[ia] : -3.0e38f;
          float vb = (ib < KMAX) ? o[ib] : -3.0e38f;
          bool ta = (va >= vb);
          mg[j] = ta ? va : vb;
          ia += ta; ib += !ta;
        }
#pragma unroll
        for (int j = 0; j < KMAX; ++j) t[j] = mg[j];
      }
      kth = t[kidx - 1];
    }

    // second softmax over probability VALUES (as in reference); max prob = 1/sum1
    const float m2 = inv1;
    float s2 = 0.f;
    for (int n = lane; n < len; n += 32) {
      const float sv = srow[n];
      if (sv >= kth) s2 += __expf(__expf(sv - m1) * inv1 - m2);
    }
    const float inv2 = 1.0f / wave_sum(s2);

    __bf16* prow = sP + M * PSTR;
    const bool zrow = (qg == 0);           // reference zeroes row q==0
    for (int n = lane; n < Lv32; n += 32) {
      float wgt = 0.f;
      if (!zrow && n < len) {
        const float sv = srow[n];
        if (sv >= kth) wgt = __expf(__expf(sv - m1) * inv1 - m2) * inv2;
      }
      prow[n] = (__bf16)wgt;
    }
  }
  __syncthreads();

  // ---------------- Phase 3: O = P V (bf16 WMMA over causal extent) ----------
  const int d0 = w * 16;                   // each wave owns 16 output columns
  v8f acc = {};
  for (int k0 = 0; k0 < Lv32; k0 += 32) {
    v16bf a;
    const __bf16* prow = sP + r * PSTR + k0 + half * 8;
#pragma unroll
    for (int j = 0; j < 8; ++j) a[j]     = prow[j];
#pragma unroll
    for (int j = 0; j < 8; ++j) a[8 + j] = prow[16 + j];

    v16bf b;   // column d0+r of V, K-chunk split by lane-half
    const float* vcol = Vp + (size_t)(k0 + half * 8) * DKD + d0 + r;
#pragma unroll
    for (int j = 0; j < 16; ++j) b[j] = (__bf16)vcol[(j < 8 ? j : j + 8) * DKD];

    acc = __builtin_amdgcn_wmma_f32_16x16x32_bf16(
        false, a, false, b, (short)0, acc, false, false);
  }
#pragma unroll
  for (int i = 0; i < 8; ++i) {
    const int M = i + 8 * half;
    Op[(size_t)(q0 + M) * DKD + d0 + r] = acc[i];
  }
}

extern "C" void kernel_launch(void* const* d_in, const int* in_sizes, int n_in,
                              void* d_out, int out_size, void* d_ws, size_t ws_size,
                              hipStream_t stream) {
  (void)in_sizes; (void)n_in; (void)out_size; (void)d_ws; (void)ws_size;
  const float* Q    = (const float*)d_in[0];
  const float* K    = (const float*)d_in[1];
  const float* V    = (const float*)d_in[2];
  /* d_in[3] = causal mask, computed analytically in-kernel */
  const int*   kidx = (const int*)d_in[4];
  float*       O    = (float*)d_out;

  const size_t shmem = (size_t)16 * SEQ * sizeof(float)      // f32 score slab
                     + (size_t)16 * PSTR * sizeof(__bf16);   // bf16 prob slab
  (void)hipFuncSetAttribute((const void*)topk_causal_attn,
                            hipFuncAttributeMaxDynamicSharedMemorySize,
                            (int)shmem);

  dim3 grid(SEQ / 16, BSZ * NH);
  topk_causal_attn<<<grid, 256, shmem, stream>>>(Q, K, V, kidx, O);
}
